// TransformerBlockEncoder_68427418960338
// MI455X (gfx1250) — compile-verified, matrix operand order checked by reading, not run on hardware
//
#include <hip/hip_runtime.h>
#include <hip/hip_bf16.h>
#include <math.h>

// ---------------- problem constants ----------------
#define B_      2
#define C_      192
#define H_      192
#define W_      192
#define HW_     36864
#define HEADS_  4
#define HD_     48
#define C3_     576        // 3*C
#define HID_    510        // int(192*2.66)
#define HID2_   1020       // 2*HID
#define KPAD_   512        // padded K for ffn_out GEMM

#define NKC_    64         // K-split chunks for gram matrices
#define NCHUNK_ 72         // pixel chunks for l2-norm reduction
#define CHUNKP_ 512        // pixels per chunk (72*512 == 36864)

typedef __bf16 bf16;
typedef __attribute__((ext_vector_type(8)))  bf16  v8bf;
typedef __attribute__((ext_vector_type(16))) bf16  v16bf;
typedef __attribute__((ext_vector_type(8)))  float v8f;
typedef __attribute__((ext_vector_type(4)))  int   v4i;

#if defined(__has_builtin)
#if __has_builtin(__builtin_amdgcn_global_load_async_to_lds_b128) && \
    __has_builtin(__builtin_amdgcn_s_wait_asynccnt)
#define USE_ASYNC_LDS 1
#endif
#endif
#ifndef USE_ASYNC_LDS
#define USE_ASYNC_LDS 0
#endif

__device__ __forceinline__ bf16 f2bf(float x) { return (bf16)x; }

__device__ __forceinline__ v8f v8f_zero() {
  v8f z;
#pragma unroll
  for (int i = 0; i < 8; ++i) z[i] = 0.0f;
  return z;
}

// A/B fragment loader for v_wmma_f32_16x16x32_bf16.
// "row" points at (this lane's row, k=0) of a row-major K-contiguous tile.
// Per ISA 7.12.2: lane L holds row L%16; kh=L/16 selects K-halves;
// VGPR j (j<4): k = 2j + 8*kh ; VGPR 4+j: k = 16 + 2j + 8*kh.
__device__ __forceinline__ v16bf frag_row(const bf16* row, int kh) {
  v8bf lo = *(const v8bf*)(row + 8 * kh);
  v8bf hi = *(const v8bf*)(row + 16 + 8 * kh);
  return __builtin_shufflevector(lo, hi, 0, 1, 2, 3, 4, 5, 6, 7,
                                 8, 9, 10, 11, 12, 13, 14, 15);
}

__device__ __forceinline__ v8f wmma_bf16(v16bf a, v16bf b, v8f c) {
  return __builtin_amdgcn_wmma_f32_16x16x32_bf16(false, a, false, b,
                                                 (short)0, c, false, false);
}

// 16B global -> LDS copy, async on CDNA5 when the builtin exists.
__device__ __forceinline__ void cp16_g2s(bf16* l, const bf16* g) {
#if USE_ASYNC_LDS
  __builtin_amdgcn_global_load_async_to_lds_b128(
      (__attribute__((address_space(1))) v4i*)g,
      (__attribute__((address_space(3))) v4i*)l, 0, 0);
#else
  *(v8bf*)l = *(const v8bf*)g;
#endif
}

__device__ __forceinline__ void wait_async_fills() {
#if USE_ASYNC_LDS
  __builtin_amdgcn_s_wait_asynccnt(0);
#endif
}

// ---------------- weight convert (+K pad) ----------------
// src fp32 [N][K] -> dst bf16 [N][KP], zero padded
__global__ void k_cvt_w(const float* __restrict__ src, bf16* __restrict__ dst,
                        int N, int K, int KP) {
  long idx = (long)blockIdx.x * blockDim.x + threadIdx.x;
  long total = (long)N * KP;
  if (idx >= total) return;
  int k = (int)(idx % KP);
  long n = idx / KP;
  dst[idx] = f2bf(k < K ? src[n * K + k] : 0.0f);
}

// ---------------- layernorm (channel-major fp32 in, pixel-major bf16 out) ---
__global__ void k_layernorm(const float* __restrict__ x,
                            const float* __restrict__ g,
                            const float* __restrict__ bta,
                            bf16* __restrict__ y) {
  long idx = (long)blockIdx.x * blockDim.x + threadIdx.x;  // over B*HW
  if (idx >= (long)B_ * HW_) return;
  int b = (int)(idx / HW_);
  int p = (int)(idx % HW_);
  const float* xb = x + (long)b * C_ * HW_ + p;
  float s = 0.f, s2 = 0.f;
  for (int c = 0; c < C_; ++c) {
    float v = xb[(long)c * HW_];
    s += v;
    s2 += v * v;
  }
  float mu = s * (1.0f / C_);
  float var = s2 * (1.0f / C_) - mu * mu;
  float inv = rsqrtf(var + 1e-5f);
  bf16* yr = y + idx * C_;
  for (int c = 0; c < C_; ++c) {
    float v = xb[(long)c * HW_];
    yr[c] = f2bf((v - mu) * inv * g[c] + bta[c]);
  }
}

// ---------------- main WMMA GEMM ----------------
// Out[m,n] = sum_k X[m,k] * Wt[n,k]     (m = pixel, n = out channel)
// X bf16 pixel-major [M][K], Wt bf16 [N][K] (both K-contiguous -> A-style loads)
// mode 0: Out fp32 pixel-major [M][N]
// mode 1: Out fp32 channel-major [N][HW] with channel-major residual add
#define GBM 64
#define GBN 64
#define GBK 32
#define LDSK 40  // padded LDS row stride (bf16 elems)

__global__ __launch_bounds__(128) void k_gemm(
    const bf16* __restrict__ X, const bf16* __restrict__ Wt,
    float* __restrict__ Out, const float* __restrict__ Res,
    int N, int K, long xBatch, long oBatch, int mode) {
  __shared__ __align__(16) bf16 As[GBM * LDSK];
  __shared__ __align__(16) bf16 Bs[GBN * LDSK];
  const int b = blockIdx.z;
  const int m0 = blockIdx.y * GBM;
  const int n0 = blockIdx.x * GBN;
  const bf16* Xb = X + (long)b * xBatch;
  const int lane = threadIdx.x & 31;
  const int wave = threadIdx.x >> 5;
  v8f acc[4];
#pragma unroll
  for (int i = 0; i < 4; ++i) acc[i] = v8f_zero();

  const int nk = (K + GBK - 1) / GBK;
  for (int kt = 0; kt < nk; ++kt) {
    const int k0 = kt * GBK;
    // fill A tile: 64 rows x 32 k (8-elem chunks)
    for (int c = threadIdx.x; c < (GBM * GBK / 8); c += blockDim.x) {
      int row = c >> 2, ko = (c & 3) * 8;
      bf16* dst = &As[row * LDSK + ko];
      int gk = k0 + ko;
      const bf16* src = Xb + (long)(m0 + row) * K + gk;
      if (gk + 8 <= K) {
        cp16_g2s(dst, src);
      } else {
#pragma unroll
        for (int j = 0; j < 8; ++j) dst[j] = (gk + j < K) ? src[j] : f2bf(0.f);
      }
    }
    // fill B tile (weights): 64 rows x 32 k
    for (int c = threadIdx.x; c < (GBN * GBK / 8); c += blockDim.x) {
      int row = c >> 2, ko = (c & 3) * 8;
      bf16* dst = &Bs[row * LDSK + ko];
      int gn = n0 + row, gk = k0 + ko;
      if (gn < N && gk + 8 <= K) {
        cp16_g2s(dst, Wt + (long)gn * K + gk);
      } else {
        const bf16* src = Wt + (long)gn * K + gk;
#pragma unroll
        for (int j = 0; j < 8; ++j)
          dst[j] = (gn < N && gk + j < K) ? src[j] : f2bf(0.f);
      }
    }
    if (kt + 1 < nk)  // hint L2 for next A tile
      __builtin_prefetch(Xb + (long)(m0 + (threadIdx.x >> 1)) * K + k0 + GBK, 0, 1);
    wait_async_fills();
    __syncthreads();

    const int kh = lane >> 4;
    v16bf af = frag_row(&As[(wave * 16 + (lane & 15)) * LDSK], kh);
#pragma unroll
    for (int nt = 0; nt < 4; ++nt) {
      v16bf bfv = frag_row(&Bs[(nt * 16 + (lane & 15)) * LDSK], kh);
      acc[nt] = wmma_bf16(af, bfv, acc[nt]);
    }
    __syncthreads();
  }

  // epilogue; D layout: lane -> N col, m = r + 8*(lane/16)
  const int nLane = lane & 15;
  const int mBase = m0 + wave * 16 + 8 * (lane >> 4);
#pragma unroll
  for (int nt = 0; nt < 4; ++nt) {
    int n = n0 + nt * 16 + nLane;
    if (n >= N) continue;
#pragma unroll
    for (int r = 0; r < 8; ++r) {
      int m = mBase + r;
      float v = acc[nt][r];
      if (mode == 0) {
        Out[(long)b * oBatch + (long)m * N + n] = v;
      } else {
        long a = (long)b * oBatch + (long)n * HW_ + m;
        float rs = Res ? Res[a] : 0.0f;
        Out[a] = v + rs;
      }
    }
  }
}

// ---------------- depthwise 3x3, pixel-major ----------------
__global__ void k_dwconv3(const float* __restrict__ x,
                          const float* __restrict__ w,
                          float* __restrict__ y, int CH) {
  long idx = (long)blockIdx.x * blockDim.x + threadIdx.x;
  long total = (long)B_ * HW_ * CH;
  if (idx >= total) return;
  int ch = (int)(idx % CH);
  long bp = idx / CH;
  int p = (int)(bp % HW_);
  int b = (int)(bp / HW_);
  int hh = p / W_, ww = p % W_;
  float acc = 0.f;
#pragma unroll
  for (int dh = -1; dh <= 1; ++dh) {
    int h2 = hh + dh;
    if (h2 < 0 || h2 >= H_) continue;
#pragma unroll
    for (int dw = -1; dw <= 1; ++dw) {
      int w2 = ww + dw;
      if (w2 < 0 || w2 >= W_) continue;
      acc += w[ch * 9 + (dh + 1) * 3 + (dw + 1)] *
             x[((long)b * HW_ + (long)h2 * W_ + w2) * CH + ch];
    }
  }
  y[idx] = acc;
}

// ---------------- l2-norm: deterministic chunked sum of squares ------------
__global__ __launch_bounds__(C3_) void k_sumsq(const float* __restrict__ x,
                                               float* __restrict__ part) {
  int ch = threadIdx.x;             // 0..575
  int chunk = blockIdx.x;           // 0..NCHUNK-1
  int b = blockIdx.y;
  const float* xb = x + ((long)b * HW_ + (long)chunk * CHUNKP_) * C3_ + ch;
  float s = 0.f;
  for (int i = 0; i < CHUNKP_; ++i) {
    float v = xb[(long)i * C3_];
    s += v * v;
  }
  part[((long)b * NCHUNK_ + chunk) * C3_ + ch] = s;
}

__global__ void k_norm_final(const float* __restrict__ part,
                             float* __restrict__ norm) {
  int idx = blockIdx.x * blockDim.x + threadIdx.x;  // B*C3
  if (idx >= B_ * C3_) return;
  int b = idx / C3_, ch = idx % C3_;
  float s = 0.f;
  for (int c = 0; c < NCHUNK_; ++c) s += part[((long)b * NCHUNK_ + c) * C3_ + ch];
  float n = sqrtf(s);
  norm[idx] = fmaxf(n, 1e-12f);
}

// l2-normalize q/k, mix k/v across streams, emit bf16 qkv for both streams
__global__ void k_l2mix(const float* __restrict__ xi, const float* __restrict__ xe,
                        const float* __restrict__ ni, const float* __restrict__ ne,
                        const float* __restrict__ alpha1,
                        const float* __restrict__ alpha2,
                        bf16* __restrict__ qi, bf16* __restrict__ qe) {
  long idx = (long)blockIdx.x * blockDim.x + threadIdx.x;
  long total = (long)B_ * HW_ * C3_;
  if (idx >= total) return;
  int ch = (int)(idx % C3_);
  int b = (int)(idx / ((long)HW_ * C3_));
  float vi = xi[idx], ve = xe[idx];
  float oi, oe;
  if (ch < C_) {                       // q: l2norm each stream
    oi = vi / ni[b * C3_ + ch];
    oe = ve / ne[b * C3_ + ch];
  } else if (ch < 2 * C_) {            // k: k = l2n(ke)*a1 + l2n(k)
    float ki = vi / ni[b * C3_ + ch];
    float ke = ve / ne[b * C3_ + ch];
    oi = ke * (*alpha1) + ki;
    oe = ke;
  } else {                             // v: v = ve*a2 + v
    oi = ve * (*alpha2) + vi;
    oe = ve;
  }
  qi[idx] = f2bf(oi);
  qe[idx] = f2bf(oe);
}

// ---------------- channel gram: logits[c,d] = sum_n q[n,c] k[n,d] ----------
// K-split over NKC chunks for occupancy; LDS-transposed staging of 32x16 tiles
__global__ __launch_bounds__(32) void k_gram(const bf16* __restrict__ qkv,
                                             float* __restrict__ part) {
  __shared__ __align__(16) bf16 As[16 * LDSK];
  __shared__ __align__(16) bf16 Bs[16 * LDSK];
  int tile = blockIdx.x % 9;
  int chunk = blockIdx.x / 9;
  int g = blockIdx.y;                 // b*HEADS + h
  int b = g / HEADS_, h = g % HEADS_;
  int ct = tile / 3, dt = tile % 3;
  int cq = h * HD_ + ct * 16;         // q channel base
  int ck = C_ + h * HD_ + dt * 16;    // k channel base
  int lane = threadIdx.x;
  v8f acc = v8f_zero();
  const int kBase = chunk * (HW_ / NKC_);
  const int kSteps = (HW_ / NKC_) / 32;
  for (int s = 0; s < kSteps; ++s) {
    int p0 = kBase + s * 32;
    const bf16* rq = qkv + ((long)b * HW_ + p0 + lane) * C3_ + cq;
    const bf16* rk = qkv + ((long)b * HW_ + p0 + lane) * C3_ + ck;
    v8bf q0 = *(const v8bf*)rq;
    v8bf q1 = *(const v8bf*)(rq + 8);
    v8bf kk0 = *(const v8bf*)rk;
    v8bf kk1 = *(const v8bf*)(rk + 8);
    __syncthreads();
#pragma unroll
    for (int j = 0; j < 8; ++j) {     // transpose into LDS: [c][k]
      As[j * LDSK + lane] = q0[j];
      As[(j + 8) * LDSK + lane] = q1[j];
      Bs[j * LDSK + lane] = kk0[j];
      Bs[(j + 8) * LDSK + lane] = kk1[j];
    }
    __syncthreads();
    int kh = lane >> 4;
    v16bf af = frag_row(&As[(lane & 15) * LDSK], kh);
    v16bf bfv = frag_row(&Bs[(lane & 15) * LDSK], kh);
    acc = wmma_bf16(af, bfv, acc);
  }
  int d = lane & 15, kh = lane >> 4;
#pragma unroll
  for (int r = 0; r < 8; ++r) {
    int c = ct * 16 + kh * 8 + r;
    int dd = dt * 16 + d;
    part[(((long)chunk * (B_ * HEADS_) + g) * 48 + c) * 48 + dd] = acc[r];
  }
}

// ---------------- K-chunk reduce + temp scale + softmax(48) -> bf16[48][64] -
__global__ __launch_bounds__(64) void k_softmax(const float* __restrict__ part,
                                                const float* __restrict__ temp,
                                                bf16* __restrict__ attn) {
  __shared__ float red[64];
  int c = blockIdx.x;                 // 0..47 (row)
  int g = blockIdx.y;                 // b*HEADS + h
  int h = g % HEADS_;
  int d = threadIdx.x;                // 0..63
  float v = -3.0e38f;
  if (d < 48) {
    float s = 0.f;
    for (int ch = 0; ch < NKC_; ++ch)
      s += part[(((long)ch * (B_ * HEADS_) + g) * 48 + c) * 48 + d];
    v = s * temp[h];
  }
  red[d] = v;
  __syncthreads();
  for (int s = 32; s > 0; s >>= 1) {
    if (d < s) red[d] = fmaxf(red[d], red[d + s]);
    __syncthreads();
  }
  float mx = red[0];
  __syncthreads();
  float e = (d < 48) ? __expf(v - mx) : 0.0f;
  red[d] = e;
  __syncthreads();
  for (int s = 32; s > 0; s >>= 1) {
    if (d < s) red[d] += red[d + s];
    __syncthreads();
  }
  float sum = red[0];
  attn[((long)g * 48 + c) * 64 + d] = f2bf(e / sum);  // zero-padded to K=64
}

// ---------------- out[n,c] = sum_d v[n,d] * attn[c,d]  (WMMA, K=64) --------
__global__ __launch_bounds__(32) void k_attnout(const bf16* __restrict__ qkv,
                                                const bf16* __restrict__ attn,
                                                bf16* __restrict__ out) {
  int g = blockIdx.y;
  int b = g / HEADS_, h = g % HEADS_;
  int m0 = blockIdx.x * 16;           // pixel tile
  int lane = threadIdx.x;
  int cv = 2 * C_ + h * HD_;          // v channel base
  v8f acc[3];
#pragma unroll
  for (int i = 0; i < 3; ++i) acc[i] = v8f_zero();
  const int kh = lane >> 4;
#pragma unroll
  for (int ks = 0; ks < 2; ++ks) {
    const bf16* arow = qkv + ((long)b * HW_ + m0 + (lane & 15)) * C3_ + cv + ks * 32;
    v16bf af = frag_row(arow, kh);
#pragma unroll
    for (int nt = 0; nt < 3; ++nt) {
      const bf16* brow = attn + ((long)g * 48 + nt * 16 + (lane & 15)) * 64 + ks * 32;
      v16bf bfv = frag_row(brow, kh);
      acc[nt] = wmma_bf16(af, bfv, acc[nt]);
    }
  }
#pragma unroll
  for (int nt = 0; nt < 3; ++nt) {
    int c = h * HD_ + nt * 16 + (lane & 15);
#pragma unroll
    for (int r = 0; r < 8; ++r) {
      int m = m0 + kh * 8 + r;
      out[((long)b * HW_ + m) * C_ + c] = f2bf(acc[nt][r]);
    }
  }
}

// ---------------- gated GELU: gelu(x1)*x2 -> bf16 [B][HW][512] (pad 0) -----
__global__ void k_gated(const float* __restrict__ x, bf16* __restrict__ y) {
  long idx = (long)blockIdx.x * blockDim.x + threadIdx.x;
  long total = (long)B_ * HW_ * KPAD_;
  if (idx >= total) return;
  int ch = (int)(idx % KPAD_);
  long bp = idx / KPAD_;
  float o = 0.0f;
  if (ch < HID_) {
    float x1 = x[bp * HID2_ + ch];
    float x2 = x[bp * HID2_ + HID_ + ch];
    float gl = 0.5f * x1 * (1.0f + erff(x1 * 0.70710678118654752440f));
    o = gl * x2;
  }
  y[idx] = f2bf(o);
}

// =====================================================================
extern "C" void kernel_launch(void* const* d_in, const int* in_sizes, int n_in,
                              void* d_out, int out_size, void* d_ws, size_t ws_size,
                              hipStream_t stream) {
  (void)in_sizes; (void)n_in; (void)out_size; (void)ws_size;
  // ---- inputs (setup_inputs dict order) ----
  const float* inp_img  = (const float*)d_in[0];
  const float* inp_edge = (const float*)d_in[1];
  const float* ln1_w  = (const float*)d_in[2];
  const float* ln1_b  = (const float*)d_in[3];
  const float* ln1e_w = (const float*)d_in[4];
  const float* ln1e_b = (const float*)d_in[5];
  const float* ln2_w  = (const float*)d_in[6];
  const float* ln2_b  = (const float*)d_in[7];
  const float* ln2e_w = (const float*)d_in[8];
  const float* ln2e_b = (const float*)d_in[9];
  const float* qkv_w   = (const float*)d_in[10];
  const float* qkv_dw  = (const float*)d_in[11];
  const float* qkve_w  = (const float*)d_in[12];
  const float* qkve_dw = (const float*)d_in[13];
  const float* proj_w  = (const float*)d_in[14];
  const float* proje_w = (const float*)d_in[15];
  const float* temp    = (const float*)d_in[16];
  const float* temp_e  = (const float*)d_in[17];
  const float* alpha1  = (const float*)d_in[18];
  const float* alpha2  = (const float*)d_in[19];
  const float* ffn_in_w   = (const float*)d_in[20];
  const float* ffn_dw     = (const float*)d_in[21];
  const float* ffn_out_w  = (const float*)d_in[22];
  const float* ffne_in_w  = (const float*)d_in[23];
  const float* ffne_dw    = (const float*)d_in[24];
  const float* ffne_out_w = (const float*)d_in[25];

  float* outx = (float*)d_out;
  float* oute = outx + (size_t)B_ * C_ * HW_;

  // ---- workspace bump allocator ----
  char* wp = (char*)d_ws;
  auto alloc = [&](size_t bytes) -> void* {
    void* r = (void*)wp;
    wp += (bytes + 255) & ~(size_t)255;
    return r;
  };
  bf16* wQKV  = (bf16*)alloc((size_t)C3_ * C_ * 2);
  bf16* wQKVE = (bf16*)alloc((size_t)C3_ * C_ * 2);
  bf16* wP    = (bf16*)alloc((size_t)C_ * C_ * 2);
  bf16* wPE   = (bf16*)alloc((size_t)C_ * C_ * 2);
  bf16* wFI   = (bf16*)alloc((size_t)HID2_ * C_ * 2);
  bf16* wFIE  = (bf16*)alloc((size_t)HID2_ * C_ * 2);
  bf16* wFO   = (bf16*)alloc((size_t)C_ * KPAD_ * 2);
  bf16* wFOE  = (bf16*)alloc((size_t)C_ * KPAD_ * 2);
  bf16* XLN = (bf16*)alloc((size_t)B_ * HW_ * C_ * 2);
  bf16* ELN = (bf16*)alloc((size_t)B_ * HW_ * C_ * 2);
  float* BUFA = (float*)alloc((size_t)B_ * HW_ * HID2_ * 4);  // big fp32 scratch 1
  float* BUFB = (float*)alloc((size_t)B_ * HW_ * HID2_ * 4);  // big fp32 scratch 2
  float* BUFC = (float*)alloc((size_t)B_ * HW_ * C3_ * 4);    // qkv fp32 / attn-out bf16
  float* PARTI = (float*)alloc((size_t)B_ * NCHUNK_ * C3_ * 4);
  float* PARTE = (float*)alloc((size_t)B_ * NCHUNK_ * C3_ * 4);
  float* NORMI = (float*)alloc((size_t)B_ * C3_ * 4);
  float* NORME = (float*)alloc((size_t)B_ * C3_ * 4);
  float* GPART = (float*)alloc((size_t)NKC_ * B_ * HEADS_ * 48 * 48 * 4);
  bf16* ATTNI = (bf16*)alloc((size_t)B_ * HEADS_ * 48 * 64 * 2);
  bf16* ATTNE = (bf16*)alloc((size_t)B_ * HEADS_ * 48 * 64 * 2);
  // aliases (lifetimes disjoint):
  bf16* QI = (bf16*)BUFA;                                   // mixed qkv img (bf16)
  bf16* QE = QI + (size_t)B_ * HW_ * C3_ + 64;              // mixed qkv edge (+pad)
  bf16* OUTI = (bf16*)BUFC;                                 // attention out img
  bf16* OUTE = OUTI + (size_t)B_ * HW_ * C_;                // attention out edge
  bf16* GL = (bf16*)BUFA;                                   // gated-gelu bf16

  // ---- launch helpers ----
  auto cvt = [&](const float* s, bf16* d, int N, int K, int KP) {
    long tot = (long)N * KP;
    k_cvt_w<<<dim3((unsigned)((tot + 255) / 256)), 256, 0, stream>>>(s, d, N, K, KP);
  };
  auto ln = [&](const float* x, const float* g, const float* b, bf16* y) {
    long tot = (long)B_ * HW_;
    k_layernorm<<<dim3((unsigned)((tot + 255) / 256)), 256, 0, stream>>>(x, g, b, y);
  };
  auto gemm = [&](const bf16* X, const bf16* Wt, float* Out, const float* Res,
                  int N, int K, int mode) {
    dim3 grid((unsigned)((N + GBN - 1) / GBN), HW_ / GBM, B_);
    long xb = (long)HW_ * K;
    long ob = (long)HW_ * N;  // same stride pixel- or channel-major
    k_gemm<<<grid, 128, 0, stream>>>(X, Wt, Out, Res, N, K, xb, ob, mode);
  };
  auto dwc = [&](const float* x, const float* w, float* y, int CH) {
    long tot = (long)B_ * HW_ * CH;
    k_dwconv3<<<dim3((unsigned)((tot + 255) / 256)), 256, 0, stream>>>(x, w, y, CH);
  };

  // ---- weight conversion ----
  cvt(qkv_w,  wQKV,  C3_, C_, C_);
  cvt(qkve_w, wQKVE, C3_, C_, C_);
  cvt(proj_w,  wP,  C_, C_, C_);
  cvt(proje_w, wPE, C_, C_, C_);
  cvt(ffn_in_w,  wFI,  HID2_, C_, C_);
  cvt(ffne_in_w, wFIE, HID2_, C_, C_);
  cvt(ffn_out_w,  wFO,  C_, HID_, KPAD_);
  cvt(ffne_out_w, wFOE, C_, HID_, KPAD_);

  // ---- attention path ----
  ln(inp_img,  ln1_w,  ln1_b,  XLN);
  ln(inp_edge, ln1e_w, ln1e_b, ELN);
  gemm(XLN, wQKV, BUFA, nullptr, C3_, C_, 0);
  dwc(BUFA, qkv_dw, BUFC, C3_);                 // img qkv fp32 -> BUFC
  gemm(ELN, wQKVE, BUFA, nullptr, C3_, C_, 0);
  dwc(BUFA, qkve_dw, BUFB, C3_);                // edge qkv fp32 -> BUFB

  k_sumsq<<<dim3(NCHUNK_, B_), C3_, 0, stream>>>(BUFC, PARTI);
  k_sumsq<<<dim3(NCHUNK_, B_), C3_, 0, stream>>>(BUFB, PARTE);
  k_norm_final<<<dim3((B_ * C3_ + 255) / 256), 256, 0, stream>>>(PARTI, NORMI);
  k_norm_final<<<dim3((B_ * C3_ + 255) / 256), 256, 0, stream>>>(PARTE, NORME);
  {
    long tot = (long)B_ * HW_ * C3_;
    k_l2mix<<<dim3((unsigned)((tot + 255) / 256)), 256, 0, stream>>>(
        BUFC, BUFB, NORMI, NORME, alpha1, alpha2, QI, QE);
  }
  k_gram<<<dim3(NKC_ * 9, B_ * HEADS_), 32, 0, stream>>>(QI, GPART);
  k_softmax<<<dim3(48, B_ * HEADS_), 64, 0, stream>>>(GPART, temp, ATTNI);
  k_gram<<<dim3(NKC_ * 9, B_ * HEADS_), 32, 0, stream>>>(QE, GPART);
  k_softmax<<<dim3(48, B_ * HEADS_), 64, 0, stream>>>(GPART, temp_e, ATTNE);
  k_attnout<<<dim3(HW_ / 16, B_ * HEADS_), 32, 0, stream>>>(QI, ATTNI, OUTI);
  k_attnout<<<dim3(HW_ / 16, B_ * HEADS_), 32, 0, stream>>>(QE, ATTNE, OUTE);
  gemm(OUTI, wP,  outx, inp_img,  C_, C_, 1);   // x = inp_img + proj(out)
  gemm(OUTE, wPE, oute, inp_edge, C_, C_, 1);   // e = inp_edge + proje(out_e)

  // ---- FFN (img) ----
  ln(outx, ln2_w, ln2_b, XLN);
  gemm(XLN, wFI, BUFA, nullptr, HID2_, C_, 0);
  dwc(BUFA, ffn_dw, BUFB, HID2_);
  {
    long tot = (long)B_ * HW_ * KPAD_;
    k_gated<<<dim3((unsigned)((tot + 255) / 256)), 256, 0, stream>>>(BUFB, GL);
  }
  gemm(GL, wFO, outx, outx, C_, KPAD_, 1);

  // ---- FFN (edge) ----
  ln(oute, ln2e_w, ln2e_b, XLN);
  gemm(XLN, wFIE, BUFA, nullptr, HID2_, C_, 0);
  dwc(BUFA, ffne_dw, BUFB, HID2_);
  {
    long tot = (long)B_ * HW_ * KPAD_;
    k_gated<<<dim3((unsigned)((tot + 255) / 256)), 256, 0, stream>>>(BUFB, GL);
  }
  gemm(GL, wFOE, oute, oute, C_, KPAD_, 1);
}
